// GraphModel_84756884620022
// MI455X (gfx1250) — compile-verified, matrix operand order checked by reading, not run on hardware
//
#include <hip/hip_runtime.h>
#include <hip/hip_bf16.h>

typedef __attribute__((ext_vector_type(16))) _Float16 v16h;
typedef __attribute__((ext_vector_type(8)))  _Float16 v8h;
typedef __attribute__((ext_vector_type(8)))  float    v8f;

#define NN    100000
#define NE    1600000
#define NG    1024
#define FDIM  128     // padded feature width (EMB=100 zero-padded, HID=128)
#define EMBD  100
#define OUTC  10

// ---------------------------------------------------------------------------
// Generic zero-fill (workspace is poisoned by harness; re-zero each call).
// ---------------------------------------------------------------------------
__global__ void fill_zero(float* __restrict__ p, long n) {
    long i = (long)blockIdx.x * blockDim.x + threadIdx.x;
    if (i < n) p[i] = 0.0f;
}

// ---------------------------------------------------------------------------
// AtomEncoder: h0[n, t] = sum_{f<9} atom_emb[f, x[n,f], t], zero-padded to 128.
// One block per node, 128 threads. 428KB table is L2-resident.
// ---------------------------------------------------------------------------
__global__ void atom_encoder(const int* __restrict__ x,
                             const float* __restrict__ emb,
                             float* __restrict__ h0) {
    int n = blockIdx.x;
    int t = threadIdx.x;
    float acc = 0.0f;
    if (t < EMBD) {
#pragma unroll
        for (int f = 0; f < 9; ++f) {
            int idx = x[n * 9 + f];
            acc += emb[(f * 119 + idx) * EMBD + t];
        }
    }
    h0[(long)n * FDIM + t] = acc;   // cols 100..127 stay zero (padding)
}

// ---------------------------------------------------------------------------
// Edge scatter: one wave32 per edge. Lanes read a coalesced float4 slice of
// h[src] (512B per wave) and atomically accumulate into agg[dst].
// Prefetch the source row of an edge 64 ahead (global_prefetch_b8) to help
// hide the random-gather latency in this HBM-bound phase.
// ---------------------------------------------------------------------------
__global__ void edge_scatter(const int* __restrict__ src,
                             const int* __restrict__ dst,
                             const float* __restrict__ h,
                             float* __restrict__ agg,
                             float* __restrict__ cnt) {
    long gid = (long)blockIdx.x * blockDim.x + threadIdx.x;
    long e = gid >> 5;
    int lane = (int)(gid & 31);
    if (e >= NE) return;

    long e2 = e + 64;
    if (e2 < NE) {
        int s2 = src[e2];
        __builtin_prefetch(h + (long)s2 * FDIM + lane * 4, 0, 0);
    }

    int s = src[e];           // uniform per wave -> scalar path
    int d = dst[e];
    const float4 v = *(const float4*)(h + (long)s * FDIM + lane * 4);
    float* base = agg + (long)d * FDIM + lane * 4;
    atomicAdd(base + 0, v.x);
    atomicAdd(base + 1, v.y);
    atomicAdd(base + 2, v.z);
    atomicAdd(base + 3, v.w);
    if (lane == 0) atomicAdd(cnt + d, 1.0f);
}

// ---------------------------------------------------------------------------
// Convert the four SAGE weight matrices to f16, zero-padding K=100 -> 128,
// AND pre-swizzle into per-lane WMMA B-fragment order so the GEMM kernel can
// fetch a whole 16-half fragment with two contiguous b128 loads.
//
// Fragment layout per 128x128 matrix:
//   frag fi = (col_tile*4 + ktile)*32 + lane,  halves elem = 0..15
//   p = elem>>1, lo = elem&1
//   k   = ktile*32 + 8*(lane>>4) + (p<4 ? 2p : 16+2(p-4)) + lo
//   col = col_tile*16 + (lane&15)
// ---------------------------------------------------------------------------
__global__ void convert_weights(const float* __restrict__ W1l,
                                const float* __restrict__ W1r,
                                const float* __restrict__ W2l,
                                const float* __restrict__ W2r,
                                _Float16* __restrict__ w16) {
    int t = blockIdx.x * blockDim.x + threadIdx.x;
    if (t >= 4 * FDIM * FDIM) return;
    int m    = t >> 14;
    int r    = t & (FDIM * FDIM - 1);
    int fi   = r >> 4;            // fragment index 0..1023
    int elem = r & 15;
    int lane = fi & 31;
    int tile = fi >> 5;           // 0..31
    int ct   = tile >> 2;         // column tile 0..7
    int kt   = tile & 3;          // k tile 0..3
    int p    = elem >> 1;
    int lo   = elem & 1;
    int k    = kt * 32 + 8 * (lane >> 4) + ((p < 4) ? 2 * p : 16 + 2 * (p - 4)) + lo;
    int col  = ct * 16 + (lane & 15);

    float v;
    if      (m == 0) v = (k < EMBD) ? W1l[k * FDIM + col] : 0.0f;
    else if (m == 1) v = (k < EMBD) ? W1r[k * FDIM + col] : 0.0f;
    else if (m == 2) v = W2l[k * FDIM + col];
    else             v = W2r[k * FDIM + col];
    w16[t] = (_Float16)v;
}

// ---------------------------------------------------------------------------
// SAGEConv dense part (WMMA): out[n,:] = relu( (agg[n,:]/max(cnt,1)) @ Wl
//                                              + h[n,:] @ Wr + bias )
// Block = 256 threads = 8 waves; block owns 16 node rows; wave w owns output
// columns [16w, 16w+16).
//
// Phase 1: cooperative stage of the 16x128 mean/h tiles into LDS as f16
//          (convert + 1/cnt applied once, not once per wave).
// Phase 2: per wave, 4 K-chunks of 32; A fragments via two ds_load_b128 runs
//          per operand (ISA 16-bit A 16x32 layout: lane L holds row L&15,
//          K = kb + 8*(L>>4) + [0..7] and + 16 + [0..7]);
//          B fragments via contiguous pre-swizzled global b128 loads.
// C/D layout: VGPR j -> row = j + 8*(L>>4), col = L&15.
// ---------------------------------------------------------------------------
__global__ void sage_gemm(const float* __restrict__ agg,
                          const float* __restrict__ cnt,
                          const float* __restrict__ h,
                          const v16h* __restrict__ WlF,
                          const v16h* __restrict__ WrF,
                          const float* __restrict__ bias,
                          float* __restrict__ out,
                          int relu_flag) {
    __shared__ __align__(16) _Float16 sMean[16 * FDIM];
    __shared__ __align__(16) _Float16 sH[16 * FDIM];

    int tid = threadIdx.x;
    int n0  = blockIdx.x * 16;

    // ---- Phase 1: stage + convert (each thread: 8 contiguous K of one row)
    {
        int sr = tid >> 4;            // row 0..15
        int sk = (tid & 15) * 8;      // K offset 0..120
        float invr = 1.0f / fmaxf(cnt[n0 + sr], 1.0f);
        const float4* ap = (const float4*)(agg + (long)(n0 + sr) * FDIM + sk);
        const float4* hp = (const float4*)(h   + (long)(n0 + sr) * FDIM + sk);
        float4 a0 = ap[0], a1 = ap[1];
        float4 b0 = hp[0], b1 = hp[1];
        _Float16* dm = sMean + sr * FDIM + sk;
        _Float16* dh = sH    + sr * FDIM + sk;
        dm[0] = (_Float16)(a0.x * invr); dm[1] = (_Float16)(a0.y * invr);
        dm[2] = (_Float16)(a0.z * invr); dm[3] = (_Float16)(a0.w * invr);
        dm[4] = (_Float16)(a1.x * invr); dm[5] = (_Float16)(a1.y * invr);
        dm[6] = (_Float16)(a1.z * invr); dm[7] = (_Float16)(a1.w * invr);
        dh[0] = (_Float16)b0.x; dh[1] = (_Float16)b0.y;
        dh[2] = (_Float16)b0.z; dh[3] = (_Float16)b0.w;
        dh[4] = (_Float16)b1.x; dh[5] = (_Float16)b1.y;
        dh[6] = (_Float16)b1.z; dh[7] = (_Float16)b1.w;
    }
    __syncthreads();

    // ---- Phase 2: WMMA
    int wave = tid >> 5;              // 0..7 -> column tile
    int lane = tid & 31;
    int hsel = lane >> 4;
    int r    = lane & 15;
    int col0 = wave * 16;

    v8f acc = {0.f, 0.f, 0.f, 0.f, 0.f, 0.f, 0.f, 0.f};

#pragma unroll
    for (int kt = 0; kt < 4; ++kt) {
        int kb = kt * 32 + 8 * hsel;

        v8h m0 = *(const v8h*)(sMean + r * FDIM + kb);
        v8h m1 = *(const v8h*)(sMean + r * FDIM + kb + 16);
        v8h x0 = *(const v8h*)(sH    + r * FDIM + kb);
        v8h x1 = *(const v8h*)(sH    + r * FDIM + kb + 16);
        v16h aM = __builtin_shufflevector(m0, m1, 0, 1, 2, 3, 4, 5, 6, 7,
                                          8, 9, 10, 11, 12, 13, 14, 15);
        v16h aH = __builtin_shufflevector(x0, x1, 0, 1, 2, 3, 4, 5, 6, 7,
                                          8, 9, 10, 11, 12, 13, 14, 15);

        int fi = (wave * 4 + kt) * 32 + lane;
        v16h bL = WlF[fi];
        v16h bR = WrF[fi];

        acc = __builtin_amdgcn_wmma_f32_16x16x32_f16(
                  false, aM, false, bL, (short)0, acc, false, false);
        acc = __builtin_amdgcn_wmma_f32_16x16x32_f16(
                  false, aH, false, bR, (short)0, acc, false, false);
    }

#pragma unroll
    for (int j = 0; j < 8; ++j) {
        int orow = n0 + j + 8 * hsel;
        int ocol = col0 + r;
        float v = acc[j] + bias[ocol];
        if (relu_flag) v = fmaxf(v, 0.0f);
        out[(long)orow * FDIM + ocol] = v;
    }
}

// ---------------------------------------------------------------------------
// BatchNorm (eval stats) fused with global mean-pool accumulation.
// ---------------------------------------------------------------------------
__global__ void bn_pool(const float* __restrict__ h,
                        const int* __restrict__ batch,
                        const float* __restrict__ gamma,
                        const float* __restrict__ beta,
                        const float* __restrict__ mean,
                        const float* __restrict__ var,
                        float* __restrict__ gsum,
                        float* __restrict__ gcnt) {
    int n = blockIdx.x;
    int c = threadIdx.x;
    int g = batch[n];
    float scale = gamma[c] * rsqrtf(var[c] + 1e-5f);
    float v = (h[(long)n * FDIM + c] - mean[c]) * scale + beta[c];
    atomicAdd(&gsum[(long)g * FDIM + c], v);
    if (c == 0) atomicAdd(&gcnt[g], 1.0f);
}

// ---------------------------------------------------------------------------
// MLP head + log-softmax. One block per graph (trivial FLOPs).
// ---------------------------------------------------------------------------
__global__ void mlp_head(const float* __restrict__ gsum,
                         const float* __restrict__ gcnt,
                         const float* __restrict__ W1,
                         const float* __restrict__ b1,
                         const float* __restrict__ W2,
                         const float* __restrict__ b2,
                         float* __restrict__ out) {
    __shared__ float gg[FDIM];
    __shared__ float hid[FDIM];
    __shared__ float logits[OUTC];
    int gidx = blockIdx.x;
    int t = threadIdx.x;

    float invc = 1.0f / fmaxf(gcnt[gidx], 1.0f);
    gg[t] = gsum[(long)gidx * FDIM + t] * invc;
    __syncthreads();

    float acc = b1[t];
#pragma unroll 8
    for (int k = 0; k < FDIM; ++k) acc += gg[k] * W1[k * FDIM + t];
    hid[t] = fmaxf(acc, 0.0f);
    __syncthreads();

    if (t < OUTC) {
        float a = b2[t];
#pragma unroll 8
        for (int k = 0; k < FDIM; ++k) a += hid[k] * W2[k * OUTC + t];
        logits[t] = a;
    }
    __syncthreads();

    if (t == 0) {
        float mx = logits[0];
        for (int i = 1; i < OUTC; ++i) mx = fmaxf(mx, logits[i]);
        float s = 0.0f;
        for (int i = 0; i < OUTC; ++i) s += __expf(logits[i] - mx);
        float lse = mx + __logf(s);
        for (int i = 0; i < OUTC; ++i)
            out[(long)gidx * OUTC + i] = logits[i] - lse;
    }
}

// ---------------------------------------------------------------------------
// Launch pipeline
// ---------------------------------------------------------------------------
extern "C" void kernel_launch(void* const* d_in, const int* in_sizes, int n_in,
                              void* d_out, int out_size, void* d_ws, size_t ws_size,
                              hipStream_t stream) {
    (void)in_sizes; (void)n_in; (void)out_size; (void)ws_size;

    const int*   x        = (const int*)d_in[0];
    const int*   eidx     = (const int*)d_in[1];   // [2, NE]: row0=src, row1=dst
    const int*   batch    = (const int*)d_in[2];
    const float* atom_emb = (const float*)d_in[3];
    const float* W1l      = (const float*)d_in[4];
    const float* b1l      = (const float*)d_in[5];
    const float* W1r      = (const float*)d_in[6];
    const float* W2l      = (const float*)d_in[7];
    const float* b2l      = (const float*)d_in[8];
    const float* W2r      = (const float*)d_in[9];
    const float* bn_gamma = (const float*)d_in[10];
    const float* bn_beta  = (const float*)d_in[11];
    const float* bn_mean  = (const float*)d_in[12];
    const float* bn_var   = (const float*)d_in[13];
    const float* mlpW1    = (const float*)d_in[14];
    const float* mlpb1    = (const float*)d_in[15];
    const float* mlpW2    = (const float*)d_in[16];
    const float* mlpb2    = (const float*)d_in[17];

    // Workspace carve-up (agg+cnt contiguous, gsum+gcnt contiguous for 1-shot zeroing)
    float* h0   = (float*)d_ws;                 // [NN,128]  layer input / final h
    float* h1   = h0 + (long)NN * FDIM;         // [NN,128]  layer-1 output
    float* agg  = h1 + (long)NN * FDIM;         // [NN,128]  neighbor sums
    float* cnt  = agg + (long)NN * FDIM;        // [NN]      degrees
    float* gsum = cnt + NN;                     // [NG,128]  pooled sums
    float* gcnt = gsum + (long)NG * FDIM;       // [NG]
    _Float16* w16 = (_Float16*)(gcnt + NG);     // 4 x 128x128 f16 swizzled weights

    const int* esrc = eidx;
    const int* edst = eidx + NE;

    // f16 swizzled weight staging (L2-resident for all GEMM blocks)
    convert_weights<<<(4 * FDIM * FDIM + 255) / 256, 256, 0, stream>>>(
        W1l, W1r, W2l, W2r, w16);

    // AtomEncoder -> h0 (zero-padded to 128)
    atom_encoder<<<NN, FDIM, 0, stream>>>(x, atom_emb, h0);

    const long aggcnt_n = (long)NN * FDIM + NN;
    const long pool_n   = (long)NG * FDIM + NG;
    const long escatter_threads = (long)NE * 32;
    const int  escatter_blocks  = (int)((escatter_threads + 255) / 256);

    const v16h* W1lF = (const v16h*)(w16);
    const v16h* W1rF = (const v16h*)(w16 + FDIM * FDIM);
    const v16h* W2lF = (const v16h*)(w16 + 2 * FDIM * FDIM);
    const v16h* W2rF = (const v16h*)(w16 + 3 * FDIM * FDIM);

    // ---- SAGE layer 1 ----
    fill_zero<<<(int)((aggcnt_n + 255) / 256), 256, 0, stream>>>(agg, aggcnt_n);
    edge_scatter<<<escatter_blocks, 256, 0, stream>>>(esrc, edst, h0, agg, cnt);
    sage_gemm<<<NN / 16, 256, 0, stream>>>(agg, cnt, h0, W1lF, W1rF,
                                           b1l, h1, /*relu=*/1);

    // ---- SAGE layer 2 ----
    fill_zero<<<(int)((aggcnt_n + 255) / 256), 256, 0, stream>>>(agg, aggcnt_n);
    edge_scatter<<<escatter_blocks, 256, 0, stream>>>(esrc, edst, h1, agg, cnt);
    sage_gemm<<<NN / 16, 256, 0, stream>>>(agg, cnt, h1, W2lF, W2rF,
                                           b2l, h0, /*relu=*/1);

    // ---- BN + global mean pool ----
    fill_zero<<<(int)((pool_n + 255) / 256), 256, 0, stream>>>(gsum, pool_n);
    bn_pool<<<NN, FDIM, 0, stream>>>(h0, batch, bn_gamma, bn_beta,
                                     bn_mean, bn_var, gsum, gcnt);

    // ---- MLP head + log_softmax ----
    mlp_head<<<NG, FDIM, 0, stream>>>(gsum, gcnt, mlpW1, mlpb1, mlpW2, mlpb2,
                                      (float*)d_out);
}
